// Model_39676907883508
// MI455X (gfx1250) — compile-verified, matrix operand order checked by reading, not run on hardware
//
#include <hip/hip_runtime.h>

// ---------- vector types ----------
typedef __attribute__((ext_vector_type(16))) _Float16 v16h;
typedef __attribute__((ext_vector_type(8)))  _Float16 h8;
typedef __attribute__((ext_vector_type(4)))  _Float16 h4;
typedef __attribute__((ext_vector_type(8)))  float    v8f;
typedef __attribute__((ext_vector_type(4)))  float    f4v;

#define KPAD 8
#define KROW (256 + KPAD)   // f16 elements per LDS key row (528 B: 2-way max bank pattern)

// LDS byte-offset of a generic pointer to __shared__ (addrspace(3) cast -> 32-bit offset)
typedef __attribute__((address_space(3))) char lds_char;
static __device__ __forceinline__ unsigned ldsOff(void* p) {
    return (unsigned)(unsigned long long)(lds_char*)p;
}

// =====================================================================
// Phase 1: S[a][l][b] = (query[a,l,:] . key[a,b,:]) / 16   via f16 WMMA, f32 acc
// grid = (Lc/256, 64), block = 256 (8 waves). Each wave: 32 l x 64 b tile.
// Each workgroup loops over 4 consecutive a; key[a] staged as f16 in LDS.
// =====================================================================
__global__ __launch_bounds__(256)
void attn_qk_wmma(const float* __restrict__ q,
                  const float* __restrict__ key,
                  float* __restrict__ S,
                  int l_base, int Lc)
{
    __shared__ __align__(16) _Float16 keyLDS[64 * KROW];

    const int tid  = threadIdx.x;
    const int lane = tid & 31;
    const int wave = tid >> 5;
    const int hw   = lane >> 4;    // half-wave: selects which K runs this lane holds
    const int lr   = lane & 15;    // row (A) / col (B) within 16

    const int l0 = l_base + blockIdx.x * 256 + wave * 32;   // this wave's first l

    for (int i = 0; i < 4; ++i) {
        const int a = blockIdx.y * 4 + i;

        // ---- stage key[a] (64 x 256 f32) -> f16 LDS, padded rows ----
        const f4v* ksrc = (const f4v*)(key + (size_t)a * 64 * 256);
        #pragma unroll
        for (int j = 0; j < 16; ++j) {
            int fi = j * 256 + tid;          // float4 index 0..4095
            f4v kv = ksrc[fi];
            int b  = fi >> 6;                // (fi*4) >> 8
            int k  = (fi * 4) & 255;
            *(h4*)&keyLDS[b * KROW + k] = __builtin_convertvector(kv, h4);
        }
        // prefetch next a's key slab while we compute (global_prefetch_b8)
        if (i + 1 < 4)
            __builtin_prefetch(key + ((size_t)(a + 1) * 64 * 256) + tid * 64, 0, 1);
        __syncthreads();

        v8f acc[2][4];
        #pragma unroll
        for (int mt = 0; mt < 2; ++mt)
            #pragma unroll
            for (int nt = 0; nt < 4; ++nt)
                acc[mt][nt] = (v8f){0.f,0.f,0.f,0.f,0.f,0.f,0.f,0.f};

        // K loop: 256 in steps of 32
        for (int k0 = 0; k0 < 256; k0 += 32) {
            // A fragments: direct global f32 loads + cvt, ISA 16-bit A layout:
            //   lanes 0-15  hold K = k0+{0..7} and k0+{16..23}
            //   lanes 16-31 hold K = k0+{8..15} and k0+{24..31}
            v16h afrag[2];
            #pragma unroll
            for (int mt = 0; mt < 2; ++mt) {
                const float* qr = q + ((size_t)a * 2048 + (l0 + mt * 16 + lr)) * 256
                                    + k0 + hw * 8;
                f4v q0 = *(const f4v*)(qr);
                f4v q1 = *(const f4v*)(qr + 4);
                f4v q2 = *(const f4v*)(qr + 16);
                f4v q3 = *(const f4v*)(qr + 20);
                h4 c0 = __builtin_convertvector(q0, h4);
                h4 c1 = __builtin_convertvector(q1, h4);
                h4 c2 = __builtin_convertvector(q2, h4);
                h4 c3 = __builtin_convertvector(q3, h4);
                h8 lo = __builtin_shufflevector(c0, c1, 0,1,2,3,4,5,6,7);
                h8 hi = __builtin_shufflevector(c2, c3, 0,1,2,3,4,5,6,7);
                afrag[mt] = __builtin_shufflevector(lo, hi,
                              0,1,2,3,4,5,6,7,8,9,10,11,12,13,14,15);
            }
            // B fragments from LDS (same half-wave K-run pattern, lane = b column)
            #pragma unroll
            for (int nt = 0; nt < 4; ++nt) {
                const _Float16* bb = &keyLDS[(nt * 16 + lr) * KROW + k0 + hw * 8];
                h8 b0 = *(const h8*)(bb);
                h8 b1 = *(const h8*)(bb + 16);
                v16h bfrag = __builtin_shufflevector(b0, b1,
                               0,1,2,3,4,5,6,7,8,9,10,11,12,13,14,15);
                acc[0][nt] = __builtin_amdgcn_wmma_f32_16x16x32_f16(
                    false, afrag[0], false, bfrag, (short)0, acc[0][nt], false, false);
                acc[1][nt] = __builtin_amdgcn_wmma_f32_16x16x32_f16(
                    false, afrag[1], false, bfrag, (short)0, acc[1][nt], false, false);
            }
        }

        // ---- store: C layout: VGPR r -> row r (lanes 0-15) / row 8+r (lanes 16-31)
        const int ll0 = (l0 - l_base) + hw * 8;
        #pragma unroll
        for (int mt = 0; mt < 2; ++mt)
            #pragma unroll
            for (int nt = 0; nt < 4; ++nt) {
                const int b = nt * 16 + lr;
                #pragma unroll
                for (int r = 0; r < 8; ++r) {
                    const int ll = ll0 + mt * 16 + r;
                    S[((size_t)a * Lc + ll) * 64 + b] = acc[mt][nt][r] * 0.0625f;
                }
            }
        __syncthreads();   // protect keyLDS before next a's staging
    }
}

// =====================================================================
// Phase 2: per y: softmax over a (256 values) for each z, then A @ value
// grid = (Lc/4), block = 256.
// Tile staged global->LDS with GLOBAL_LOAD_ASYNC_TO_LDS_B128 (ASYNCcnt).
// Row stride 260 floats: 16B-aligned rows for the async b128 DMA,
// conflict-free column reads (pass A), 2-way worst case row reads (pass B).
// =====================================================================
#define TROW 260

__global__ __launch_bounds__(256)
void attn_softmax_v(const float* __restrict__ S,
                    const float* __restrict__ value,
                    float* __restrict__ out,
                    int y_base, int Lc)
{
    extern __shared__ float smem[];
    float* T  = smem;               // [256][TROW]
    float* mz = T + 256 * TROW;     // [256]
    float* rz = mz + 256;           // [256]
    f4v*   V4 = (f4v*)(rz + 256);   // [256] (offset is 16B aligned)

    const int tid = threadIdx.x;
    const int yl  = blockIdx.x;          // y within chunk
    const int y   = y_base + yl;
    const size_t astride = (size_t)Lc * 64;
    const float* Sy = S + (size_t)yl * 256;

    // ---- async DMA the 256 x 256 slice straight into LDS (per-a rows are 1 KB
    // contiguous in S); 16B per lane per issue, tracked by ASYNCcnt ----
    #pragma unroll 4
    for (int it = 0; it < 64; ++it) {
        int fi = it * 256 + tid;         // float4 index 0..16383
        int a  = fi >> 6;
        int z4 = (fi & 63) * 4;
        unsigned long long gaddr =
            (unsigned long long)(const void*)(Sy + (size_t)a * astride + z4);
        unsigned dst = ldsOff(&T[a * TROW + z4]);
        asm volatile("global_load_async_to_lds_b128 %0, %1, off"
                     :: "v"(dst), "v"(gaddr) : "memory");
    }
    V4[tid] = *(const f4v*)(value + ((size_t)y * 256 + tid) * 4);
    asm volatile("s_wait_asynccnt 0" ::: "memory");
    __syncthreads();

    // pass A: z = tid, reduce over a (column reads: conflict-free)
    float m = -3.402823466e38f;
    for (int a2 = 0; a2 < 256; ++a2) m = fmaxf(m, T[a2 * TROW + tid]);
    float s = 0.f;
    for (int a2 = 0; a2 < 256; ++a2) s += __expf(T[a2 * TROW + tid] - m);
    mz[tid] = m;
    rz[tid] = 1.0f / s;
    __syncthreads();

    // pass B: x = tid, out[y,x,:] = sum_z softmax * value[y,z,:]
    f4v accv = (f4v){0.f, 0.f, 0.f, 0.f};
    const float* Trow = &T[tid * TROW];
    for (int z = 0; z < 256; ++z) {
        float p = __expf(Trow[z] - mz[z]) * rz[z];
        accv += V4[z] * p;
    }
    *(f4v*)(out + (size_t)y * 1024 + tid * 4) = accv;
}

// =====================================================================
extern "C" void kernel_launch(void* const* d_in, const int* in_sizes, int n_in,
                              void* d_out, int out_size, void* d_ws, size_t ws_size,
                              hipStream_t stream)
{
    const float* q     = (const float*)d_in[0];   // (256, 2048, 256)
    const float* key   = (const float*)d_in[1];   // (256, 64, 256)
    const float* value = (const float*)d_in[2];   // (512, 256, 4)
    float* out = (float*)d_out;                   // (64, 8, 1024) == (512,256,4) flat
    float* S   = (float*)d_ws;

    // pick largest l-chunk (multiple of 256) whose S slab fits in ws
    int Lc = 2048;
    while (Lc > 256 && (size_t)256 * (size_t)Lc * 64 * sizeof(float) > ws_size)
        Lc >>= 1;

    const size_t shmem = (size_t)(256 * TROW + 256 + 256 + 256 * 4) * sizeof(float);

    for (int l_base = 0; l_base < 2048; l_base += Lc) {
        dim3 g1(Lc / 256, 64);
        attn_qk_wmma<<<g1, 256, 0, stream>>>(q, key, S, l_base, Lc);
        dim3 g2(Lc / 4);
        attn_softmax_v<<<g2, 256, shmem, stream>>>(S, value, out, l_base / 4, Lc);
    }
}
